// RelationalNetwork_43121471652419
// MI455X (gfx1250) — compile-verified, matrix operand order checked by reading, not run on hardware
//
#include <hip/hip_runtime.h>
#include <hip/hip_bf16.h>

// ---------------------------------------------------------------------------
// Relational Network forward on gfx1250 (CDNA5).
// Heavy GEMMs: v_wmma_f32_16x16x32_f16 (f16 in, f32 acc), B panels staged in
// LDS via GLOBAL_LOAD_ASYNC_TO_LDS_B128 when available (ASYNCcnt path).
// ---------------------------------------------------------------------------

typedef __attribute__((ext_vector_type(16))) _Float16 v16h;
typedef __attribute__((ext_vector_type(8)))  float    v8f;
typedef __attribute__((ext_vector_type(2)))  _Float16 half2_t;
typedef __attribute__((ext_vector_type(4)))  unsigned int uint4v;
typedef __attribute__((ext_vector_type(4)))  int int4v;

#define B_    64
#define L_    32
#define EMB_  300
#define EMBP_ 320     // K pad (mult of 32)
#define HID_  500
#define HIDP_ 512     // K pad
#define G4_   2000    // 4*HID, already mult of 16
#define GH_   100
#define GHKP_ 128     // g2 K pad
#define GHNP_ 112     // g2 N pad (mult of 16)
#define OBJ_  64
#define CH_   24
#define ANS_  1000
#define CONCAT_ 548

#if defined(__gfx1250__) && __has_builtin(__builtin_amdgcn_global_load_async_to_lds_b128)
#define HAVE_ASYNC_LDS 1
#else
#define HAVE_ASYNC_LDS 0
#endif

#if HAVE_ASYNC_LDS
typedef __attribute__((address_space(1))) int4v* gas_p;   // global int4*
typedef __attribute__((address_space(3))) int4v* las_p;   // LDS int4*
__device__ __forceinline__ void async_wait0() {
#if __has_builtin(__builtin_amdgcn_s_wait_asynccnt)
  __builtin_amdgcn_s_wait_asynccnt(0);
#else
  asm volatile("s_wait_asynccnt 0x0" ::: "memory");
#endif
}
#endif

// ---------------- helpers: f32 -> f16 convert with zero padding ------------
__global__ void k_cvt_pad_f16(_Float16* __restrict__ dst, const float* __restrict__ src,
                              int N, int K, int Np, int Kp) {
  int idx = blockIdx.x * blockDim.x + threadIdx.x;
  if (idx >= Np * Kp) return;
  int n = idx / Kp, k = idx % Kp;
  float v = (n < N && k < K) ? src[(size_t)n * K + k] : 0.f;
  dst[idx] = (_Float16)v;
}

// ---------------- embedding gather into padded f16 X -----------------------
__global__ void k_embed_gather(_Float16* __restrict__ Xp, const int* __restrict__ sent,
                               const float* __restrict__ table) {
  int idx = blockIdx.x * blockDim.x + threadIdx.x;
  if (idx >= B_ * L_ * EMBP_) return;
  int m = idx / EMBP_, k = idx % EMBP_;          // m = b*L + t
  int tok = sent[m];
  float v = (k < EMB_) ? table[(size_t)tok * EMB_ + k] : 0.f;
  Xp[idx] = (_Float16)v;
}

// ---------------- LSTM state init ------------------------------------------
__global__ void k_init_state(float* __restrict__ h, float* __restrict__ c,
                             _Float16* __restrict__ h16,
                             const float* __restrict__ h0, const float* __restrict__ c0) {
  int idx = blockIdx.x * blockDim.x + threadIdx.x;
  if (idx >= B_ * HIDP_) return;
  int b = idx / HIDP_, k = idx % HIDP_;
  float hv = (k < HID_) ? h0[b * HID_ + k] : 0.f;
  h16[idx] = (_Float16)hv;
  if (k < HID_) {
    h[b * HID_ + k] = hv;
    c[b * HID_ + k] = c0[b * HID_ + k];
  }
}

// ---------------- WMMA GEMM:  C[M][N] = A[M][Kp] * B[N][Kp]^T --------------
// Block = 4 waves covering a 64-row M stripe of one 16-col N tile.
// The shared 16xKp B panel is staged in LDS (async when available); each wave
// reads B fragments from LDS (ds_load_b128) and A fragments from global.
__global__ void __launch_bounds__(128)
k_gemm_f16(const _Float16* __restrict__ A, const _Float16* __restrict__ Bw,
           float* __restrict__ C, int M, int N, int Kp) {
  extern __shared__ _Float16 bsh[];            // [16][Kp]
  int n0 = blockIdx.x << 4;
  int wave = threadIdx.x >> 5;
  int lane = threadIdx.x & 31;
  int m0 = ((blockIdx.y << 2) + wave) << 4;

  // stage B panel rows n0..n0+15 into LDS, 16B chunks across all 128 threads
  const _Float16* bpanel = Bw + (size_t)n0 * Kp;
  int nChunks = (Kp << 1);                     // 16*Kp halves / 8 per chunk
  for (int ch = threadIdx.x; ch < nChunks; ch += 128) {
#if HAVE_ASYNC_LDS
    __builtin_amdgcn_global_load_async_to_lds_b128(
        (gas_p)(bpanel + (ch << 3)), (las_p)(bsh + (ch << 3)), 0, 0);
#else
    *reinterpret_cast<uint4v*>(bsh + (ch << 3)) =
        *reinterpret_cast<const uint4v*>(bpanel + (ch << 3));
#endif
  }
#if HAVE_ASYNC_LDS
  async_wait0();
#endif
  __syncthreads();

  int hi = lane >> 4;                          // 0 or 1 (half-wave)
  int rl = lane & 15;
  const _Float16* arow = A + (size_t)(m0 + rl) * Kp;
  const _Float16* brow = bsh + (size_t)rl * Kp;
  v8f acc = {};
  for (int kb = 0; kb < Kp; kb += 32) {
    __builtin_prefetch(arow + kb + 64, 0, 3);
    v16h a, bf;
#pragma unroll
    for (int vg = 0; vg < 8; ++vg) {
      // ISA 7.12.2 A/B f16 layout: k = 2*(vg&3) + 8*(lane>=16) + 16*(vg>=4)
      int k = kb + 2 * (vg & 3) + (hi << 3) + ((vg >> 2) << 4);
      half2_t av = *reinterpret_cast<const half2_t*>(arow + k);
      half2_t bv = *reinterpret_cast<const half2_t*>(brow + k);   // ds_load
      a[2 * vg] = av.x;  a[2 * vg + 1] = av.y;
      bf[2 * vg] = bv.x; bf[2 * vg + 1] = bv.y;
    }
    acc = __builtin_amdgcn_wmma_f32_16x16x32_f16(false, a, false, bf,
                                                 (short)0, acc, false, false);
  }
#pragma unroll
  for (int r = 0; r < 8; ++r) {
    int m = m0 + r + (hi << 3);
    int n = n0 + rl;
    C[(size_t)m * N + n] = acc[r];
  }
}

// ---------------- LSTM cell (pointwise) ------------------------------------
__global__ void k_lstm_cell(const float* __restrict__ px, const float* __restrict__ ph,
                            const float* __restrict__ b_ih, const float* __restrict__ b_hh,
                            const int* __restrict__ lens,
                            float* __restrict__ h, float* __restrict__ c,
                            _Float16* __restrict__ h16, int t) {
  int idx = blockIdx.x * blockDim.x + threadIdx.x;
  if (idx >= B_ * HID_) return;
  int b = idx / HID_, k = idx % HID_;
  const float* pxr = px + (size_t)(b * L_ + t) * G4_;
  const float* phr = ph + (size_t)b * G4_;
  float gi = pxr[k]            + phr[k]            + b_ih[k]            + b_hh[k];
  float gf = pxr[HID_ + k]     + phr[HID_ + k]     + b_ih[HID_ + k]     + b_hh[HID_ + k];
  float gg = pxr[2 * HID_ + k] + phr[2 * HID_ + k] + b_ih[2 * HID_ + k] + b_hh[2 * HID_ + k];
  float go = pxr[3 * HID_ + k] + phr[3 * HID_ + k] + b_ih[3 * HID_ + k] + b_hh[3 * HID_ + k];
  float ig = 1.f / (1.f + __expf(-gi));
  float fg = 1.f / (1.f + __expf(-gf));
  float og = 1.f / (1.f + __expf(-go));
  float g  = tanhf(gg);
  float cn = fg * c[idx] + ig * g;
  float hn = og * tanhf(cn);
  if (t < lens[b]) {
    c[idx] = cn;
    h[idx] = hn;
    h16[b * HIDP_ + k] = (_Float16)hn;
  }
}

// ---------------- g-MLP layer-1 partial projections ------------------------
__global__ void k_qpart(float* __restrict__ qp, const float* __restrict__ h,
                        const float* __restrict__ g1w) {
  int idx = blockIdx.x * blockDim.x + threadIdx.x;
  if (idx >= B_ * GH_) return;
  int b = idx / GH_, k = idx % GH_;
  const float* wr = g1w + (size_t)k * CONCAT_;
  const float* hr = h + (size_t)b * HID_;
  float acc = 0.f;
  for (int j = 0; j < HID_; ++j) acc += hr[j] * wr[j];
  qp[idx] = acc;
}

__global__ void k_ijpart(float* __restrict__ ip, float* __restrict__ jp,
                         const float* __restrict__ conv, const float* __restrict__ g1w) {
  int idx = blockIdx.x * blockDim.x + threadIdx.x;
  if (idx >= B_ * OBJ_ * GH_) return;
  int k = idx % GH_;
  int bo = idx / GH_;
  int o = bo % OBJ_, b = bo / OBJ_;
  const float* wr = g1w + (size_t)k * CONCAT_;
  float accj = 0.f, acci = 0.f;
  for (int cc = 0; cc < CH_; ++cc) {
    float ov = conv[((size_t)(b * CH_ + cc)) * OBJ_ + o];   // obj[b][o][cc]
    accj += ov * wr[HID_ + cc];          // g1_w[:, HID:HID+C]  (j block)
    acci += ov * wr[HID_ + CH_ + cc];    // g1_w[:, HID+C:]     (i block)
  }
  jp[idx] = accj;
  ip[idx] = acci;
}

__global__ void k_zero(float* __restrict__ p, int n) {
  int idx = blockIdx.x * blockDim.x + threadIdx.x;
  if (idx < n) p[idx] = 0.f;
}

// ---------------- fused g-MLP layer1(on the fly) -> WMMA layer2 -> sum -----
// grid = (256 row-tiles, B); block = 224 threads = 7 waves (7 N-tiles of 16).
__global__ void __launch_bounds__(224)
k_g2_fused(const float* __restrict__ qp, const float* __restrict__ ip,
           const float* __restrict__ jp, const float* __restrict__ g1b,
           const _Float16* __restrict__ g2w16, const float* __restrict__ g2b,
           float* __restrict__ gsum) {
  __shared__ _Float16 at[16][GHKP_];   // 16 rows x 128 K (f16, zero-padded)
  int b = blockIdx.y;
  int rowTile = blockIdx.x;            // 0..255  (4096 rows / 16)

  for (int idx = threadIdx.x; idx < 16 * GHKP_; idx += 224) {
    int r = idx >> 7, k = idx & 127;
    float v = 0.f;
    if (k < GH_) {
      int row = rowTile * 16 + r;      // (i,j) pair index
      int i = row >> 6, j = row & 63;
      v = qp[b * GH_ + k] + ip[(b * OBJ_ + i) * GH_ + k]
        + jp[(b * OBJ_ + j) * GH_ + k] + g1b[k];
      v = v > 0.f ? v : 0.f;           // relu(h1)
    }
    at[r][k] = (_Float16)v;
  }
  __syncthreads();

  int wave = threadIdx.x >> 5;         // 0..6 -> N tile
  int lane = threadIdx.x & 31;
  int n0 = wave << 4;
  int hi = lane >> 4, rl = lane & 15;
  const _Float16* brow = g2w16 + (size_t)(n0 + rl) * GHKP_;
  v8f acc = {};
#pragma unroll
  for (int kb = 0; kb < GHKP_; kb += 32) {
    v16h a, bf;
#pragma unroll
    for (int vg = 0; vg < 8; ++vg) {
      int k = kb + 2 * (vg & 3) + (hi << 3) + ((vg >> 2) << 4);
      half2_t av = *reinterpret_cast<const half2_t*>(&at[rl][k]);   // ds_load
      half2_t bv = *reinterpret_cast<const half2_t*>(brow + k);
      a[2 * vg] = av.x;  a[2 * vg + 1] = av.y;
      bf[2 * vg] = bv.x; bf[2 * vg + 1] = bv.y;
    }
    acc = __builtin_amdgcn_wmma_f32_16x16x32_f16(false, a, false, bf,
                                                 (short)0, acc, false, false);
  }

  int n = n0 + rl;
  if (n < GH_) {
    float bias = g2b[n];
    float s = 0.f;
#pragma unroll
    for (int r = 0; r < 8; ++r) {
      float v = acc[r] + bias;         // h2 = relu(h1@g2_w^T + g2_b)
      s += (v > 0.f ? v : 0.f);
    }
    s += __shfl_xor(s, 16, 32);        // combine the two row-halves (same n)
    if (lane < 16) atomicAdd(&gsum[b * GH_ + n], s);
  }
}

// ---------------- f-MLP + log_softmax --------------------------------------
__global__ void __launch_bounds__(128)
k_fmlp(const float* __restrict__ gsum, const float* __restrict__ f1w,
       const float* __restrict__ f1b, const float* __restrict__ f2w,
       const float* __restrict__ f2b, float* __restrict__ out) {
  __shared__ float gs[GH_];
  __shared__ float f1v[GH_];
  __shared__ float f2v[ANS_];
  __shared__ float red[128];
  int b = blockIdx.x, t = threadIdx.x;

  if (t < GH_) gs[t] = gsum[b * GH_ + t];
  __syncthreads();
  if (t < GH_) {
    float acc = f1b[t];
    const float* wr = f1w + (size_t)t * GH_;
    for (int k = 0; k < GH_; ++k) acc += wr[k] * gs[k];
    f1v[t] = acc > 0.f ? acc : 0.f;
  }
  __syncthreads();
  for (int o = t; o < ANS_; o += 128) {
    float acc = f2b[o];
    const float* wr = f2w + (size_t)o * GH_;
    for (int k = 0; k < GH_; ++k) acc += wr[k] * f1v[k];
    f2v[o] = acc > 0.f ? acc : 0.f;
  }
  __syncthreads();
  float m = -1e30f;
  for (int o = t; o < ANS_; o += 128) m = fmaxf(m, f2v[o]);
  red[t] = m; __syncthreads();
  for (int s = 64; s > 0; s >>= 1) {
    if (t < s) red[t] = fmaxf(red[t], red[t + s]);
    __syncthreads();
  }
  m = red[0]; __syncthreads();
  float se = 0.f;
  for (int o = t; o < ANS_; o += 128) se += __expf(f2v[o] - m);
  red[t] = se; __syncthreads();
  for (int s = 64; s > 0; s >>= 1) {
    if (t < s) red[t] += red[t + s];
    __syncthreads();
  }
  float logZ = m + __logf(red[0]);
  for (int o = t; o < ANS_; o += 128) out[(size_t)b * ANS_ + o] = f2v[o] - logZ;
}

// ---------------------------------------------------------------------------
extern "C" void kernel_launch(void* const* d_in, const int* in_sizes, int n_in,
                              void* d_out, int out_size, void* d_ws, size_t ws_size,
                              hipStream_t stream) {
  (void)in_sizes; (void)n_in; (void)out_size; (void)ws_size;
  const int*   sent  = (const int*)d_in[0];
  const float* conv  = (const float*)d_in[1];
  const int*   lens  = (const int*)d_in[2];
  const float* table = (const float*)d_in[3];
  const float* W_ih  = (const float*)d_in[4];
  const float* W_hh  = (const float*)d_in[5];
  const float* b_ih  = (const float*)d_in[6];
  const float* b_hh  = (const float*)d_in[7];
  const float* h0    = (const float*)d_in[8];
  const float* c0    = (const float*)d_in[9];
  const float* g1w   = (const float*)d_in[10];
  const float* g1b   = (const float*)d_in[11];
  const float* g2w   = (const float*)d_in[12];
  const float* g2b   = (const float*)d_in[13];
  const float* f1w   = (const float*)d_in[14];
  const float* f1b   = (const float*)d_in[15];
  const float* f2w   = (const float*)d_in[16];
  const float* f2b   = (const float*)d_in[17];
  float* out = (float*)d_out;

  // workspace carve-up (256B aligned)
  char* p = (char*)d_ws;
  auto alloc = [&](size_t bytes) -> char* {
    char* r = p;
    p += (bytes + 255) & ~(size_t)255;
    return r;
  };
  _Float16* Xp    = (_Float16*)alloc((size_t)B_ * L_ * EMBP_ * 2);
  _Float16* Wih16 = (_Float16*)alloc((size_t)G4_ * EMBP_ * 2);
  _Float16* Whh16 = (_Float16*)alloc((size_t)G4_ * HIDP_ * 2);
  _Float16* g2w16 = (_Float16*)alloc((size_t)GHNP_ * GHKP_ * 2);
  float* px    = (float*)alloc((size_t)B_ * L_ * G4_ * 4);
  float* ph    = (float*)alloc((size_t)B_ * G4_ * 4);
  float* h     = (float*)alloc((size_t)B_ * HID_ * 4);
  float* c     = (float*)alloc((size_t)B_ * HID_ * 4);
  _Float16* h16 = (_Float16*)alloc((size_t)B_ * HIDP_ * 2);
  float* qp    = (float*)alloc((size_t)B_ * GH_ * 4);
  float* ipart = (float*)alloc((size_t)B_ * OBJ_ * GH_ * 4);
  float* jpart = (float*)alloc((size_t)B_ * OBJ_ * GH_ * 4);
  float* gsum  = (float*)alloc((size_t)B_ * GH_ * 4);

  const int thr = 256;
  // weight conversion (once per call; deterministic)
  k_cvt_pad_f16<<<(G4_ * EMBP_ + thr - 1) / thr, thr, 0, stream>>>(Wih16, W_ih, G4_, EMB_, G4_, EMBP_);
  k_cvt_pad_f16<<<(G4_ * HIDP_ + thr - 1) / thr, thr, 0, stream>>>(Whh16, W_hh, G4_, HID_, G4_, HIDP_);
  k_cvt_pad_f16<<<(GHNP_ * GHKP_ + thr - 1) / thr, thr, 0, stream>>>(g2w16, g2w, GH_, GH_, GHNP_, GHKP_);
  k_embed_gather<<<(B_ * L_ * EMBP_ + thr - 1) / thr, thr, 0, stream>>>(Xp, sent, table);
  k_init_state<<<(B_ * HIDP_ + thr - 1) / thr, thr, 0, stream>>>(h, c, h16, h0, c0);

  // hoisted input-side LSTM GEMM: px[b*L+t][2000] = X @ W_ih^T
  k_gemm_f16<<<dim3(G4_ / 16, (B_ * L_) / 64), 128, 16 * EMBP_ * 2, stream>>>(
      Xp, Wih16, px, B_ * L_, G4_, EMBP_);

  // sequential recurrence: ph = h @ W_hh^T each step
  for (int t = 0; t < L_; ++t) {
    k_gemm_f16<<<dim3(G4_ / 16, 1), 128, 16 * HIDP_ * 2, stream>>>(
        h16, Whh16, ph, B_, G4_, HIDP_);
    k_lstm_cell<<<(B_ * HID_ + thr - 1) / thr, thr, 0, stream>>>(px, ph, b_ih, b_hh, lens, h, c, h16, t);
  }

  // relational module
  k_qpart<<<(B_ * GH_ + thr - 1) / thr, thr, 0, stream>>>(qp, h, g1w);
  k_ijpart<<<(B_ * OBJ_ * GH_ + thr - 1) / thr, thr, 0, stream>>>(ipart, jpart, conv, g1w);
  k_zero<<<(B_ * GH_ + thr - 1) / thr, thr, 0, stream>>>(gsum, B_ * GH_);
  k_g2_fused<<<dim3((OBJ_ * OBJ_) / 16, B_), 224, 0, stream>>>(qp, ipart, jpart, g1b, g2w16, g2b, gsum);

  // head
  k_fmlp<<<B_, 128, 0, stream>>>(gsum, f1w, f1b, f2w, f2b, out);
}
